// MultiheadAttention_25709674234507
// MI455X (gfx1250) — matrix operand order repaired and checked by reading, not compiled, not run on hardware
//
#include <hip/hip_runtime.h>
#include <hip/hip_bf16.h>

// Problem constants (from reference): B=8, S=1024, D=1024, H=16, DK=64
static constexpr int Bc = 8;
static constexpr int Sc = 1024;
static constexpr int Dc = 1024;
static constexpr int Hc = 16;
static constexpr int DKc = 64;

typedef __attribute__((ext_vector_type(16))) __bf16 v16bf;
typedef __attribute__((ext_vector_type(8)))  float  v8f;
typedef __attribute__((ext_vector_type(4)))  unsigned int uint32x4;
typedef __attribute__((ext_vector_type(8)))  int          int32x8;
typedef __attribute__((ext_vector_type(4)))  int          int32x4;

#if __has_builtin(__builtin_amdgcn_tensor_load_to_lds)
#define HAVE_TDM 1
#else
#define HAVE_TDM 0
#endif

// ---------------------------------------------------------------------------
// Helpers
// ---------------------------------------------------------------------------
struct BF16Frag { union { v16bf v; unsigned int u[8]; }; };

__device__ inline unsigned pack_bf16(float a, float b) {
  union { __bf16 h[2]; unsigned u; } t;
  t.h[0] = (__bf16)a; t.h[1] = (__bf16)b;   // -> v_cvt_pk_bf16_f32
  return t.u;
}

// A (16x32 MxK) fragment: lane row = lane&15, half = lane>>4.
//  VGPR i: k = 16*half + 2i
__device__ inline v16bf load_a_frag(const __bf16* rowbase, int half) {
  const unsigned int* p = (const unsigned int*)rowbase; // 4B aligned
  BF16Frag f;
#pragma unroll
  for (int i = 0; i < 8; ++i) {
    int k0 = 16 * half + 2 * i;
    f.u[i] = p[k0 >> 1];
  }
  return f.v;
}

// B (32x16 KxN) fragment: lane col = lane&15, half = lane>>4; VGPR i: k = 16*half + 2i
__device__ inline v16bf load_b_frag(const __bf16* colbase, int half) {
  const unsigned int* p = (const unsigned int*)colbase;
  BF16Frag f;
#pragma unroll
  for (int i = 0; i < 8; ++i) f.u[i] = p[8 * half + i];
  return f.v;
}

__device__ inline void wait_asynccnt0() {
#if __has_builtin(__builtin_amdgcn_s_wait_asynccnt)
  __builtin_amdgcn_s_wait_asynccnt(0);
#else
  asm volatile("s_wait_asynccnt 0" ::: "memory");
#endif
}
__device__ inline void wait_asynccnt4() {
#if __has_builtin(__builtin_amdgcn_s_wait_asynccnt)
  __builtin_amdgcn_s_wait_asynccnt(4);
#else
  asm volatile("s_wait_asynccnt 4" ::: "memory");
#endif
}

#if HAVE_TDM
// TDM: load a 2D bf16 tile [tile_d1 rows x tile_d0 elems], row stride
// stride_d0 elems, into LDS at lds_off with 4-DWORD padding every 32 DWORDs
// (-> LDS row pitch = tile_d0 + 8 bf16). D# bitfields per ISA 8.3/8.4.
// This toolchain exposes the 6-arg builtin:
//   (uint32x4 g0, int32x8 g1, int32x4 g2, int32x4 g3, int32x8 extra, i32 cpol)
__device__ inline void tdm_load_2d_bf16(unsigned lds_off, const __bf16* gaddr,
                                        unsigned tensor_d0, unsigned tensor_d1,
                                        unsigned stride_d0,
                                        unsigned tile_d0, unsigned tile_d1) {
  unsigned long long ga = (unsigned long long)(size_t)gaddr;
  uint32x4 g0 = {};
  g0[0] = 1u;                                           // count=1, user mode
  g0[1] = lds_off;                                      // lds_addr (bytes)
  g0[2] = (unsigned)(ga & 0xffffffffu);                 // global_addr[31:0]
  g0[3] = (unsigned)((ga >> 32) & 0x01ffffffu) | (2u << 30); // addr[56:32]|type=2
  int32x8 g1 = {};
  g1[0] = (int)((1u << 16) |                            // data_size = 2 bytes
                (1u << 20) |                            // pad_enable
                (4u << 22) |                            // pad_interval: 32 DWORDs
                (3u << 25));                            // pad_amount:   4 DWORDs
  g1[1] = (int)((tensor_d0 & 0xffffu) << 16);           // tensor_dim0 lo
  g1[2] = (int)(((tensor_d0 >> 16) & 0xffffu) | ((tensor_d1 & 0xffffu) << 16));
  g1[3] = (int)(((tensor_d1 >> 16) & 0xffffu) | ((tile_d0 & 0xffffu) << 16));
  g1[4] = (int)(tile_d1 & 0xffffu);                     // tile_dim1, tile_dim2=0
  g1[5] = (int)stride_d0;                               // tensor_dim0_stride lo
  g1[6] = 0;
  g1[7] = 0;
  int32x4 z4 = {};
  int32x8 z8 = {};
  __builtin_amdgcn_tensor_load_to_lds(g0, g1, z4, z4, z8, 0);
}
#endif

// ---------------------------------------------------------------------------
// Kernel 1/5: C[M,N] = X[M,K] @ W[N,K]^T + bias[N] (+ residual[M,N])
// Block tile 128x128, K-step 32, 8 waves; wave w owns rows w*16 and all 128
// cols (8 fragments). OUT_BF16 selects bf16 (q/k/v) vs fp32 (final) epilogue.
// ---------------------------------------------------------------------------
template <bool OUT_BF16>
__global__ __launch_bounds__(256)
void proj_gemm_kernel(const float* __restrict__ X, const float* __restrict__ W,
                      const float* __restrict__ bias,
                      const float* __restrict__ residual,
                      void* __restrict__ Cout, int M, int N, int K) {
  constexpr int TP = 40;
  __shared__ __bf16 sA[128 * TP];
  __shared__ __bf16 sB[128 * TP];

  const int tid  = threadIdx.x;
  const int wave = tid >> 5;
  const int lane = tid & 31;
  const int col  = lane & 15;
  const int half = lane >> 4;
  const int m0 = blockIdx.y * 128;
  const int n0 = blockIdx.x * 128;

  v8f acc[8] = {};

  for (int kk = 0; kk < K; kk += 32) {
#pragma unroll
    for (int i = 0; i < 4; ++i) {
      int idx = i * 256 + tid;
      int r = idx >> 3, c4 = (idx & 7) * 4;
      float4 xa = *(const float4*)&X[(size_t)(m0 + r) * K + kk + c4];
      float4 xb = *(const float4*)&W[(size_t)(n0 + r) * K + kk + c4];
      uint2 pa; pa.x = pack_bf16(xa.x, xa.y); pa.y = pack_bf16(xa.z, xa.w);
      uint2 pb; pb.x = pack_bf16(xb.x, xb.y); pb.y = pack_bf16(xb.z, xb.w);
      *(uint2*)&sA[r * TP + c4] = pa;
      *(uint2*)&sB[r * TP + c4] = pb;
      if (kk + 32 < K) {
        __builtin_prefetch(&X[(size_t)(m0 + r) * K + kk + 32 + c4], 0, 1);
        __builtin_prefetch(&W[(size_t)(n0 + r) * K + kk + 32 + c4], 0, 1);
      }
    }
    __syncthreads();

    v16bf a = load_a_frag(sA + (wave * 16 + (lane & 15)) * TP, half);
    v16bf bcur = load_b_frag(sB + col * TP, half);
#pragma unroll
    for (int nf = 0; nf < 8; ++nf) {
      v16bf bnxt = bcur;
      if (nf < 7) bnxt = load_b_frag(sB + ((nf + 1) * 16 + col) * TP, half);
      acc[nf] = __builtin_amdgcn_wmma_f32_16x16x32_bf16(
          false, a, false, bcur, (short)0, acc[nf], false, false);
      bcur = bnxt;
    }
    __syncthreads();
  }

  const int mrow = m0 + wave * 16;
#pragma unroll
  for (int nf = 0; nf < 8; ++nf) {
    int gc = n0 + nf * 16 + col;
    float bv = bias[gc];
#pragma unroll
    for (int vv = 0; vv < 8; ++vv) {
      int gr = mrow + vv + 8 * half;
      float val = acc[nf][vv] + bv;
      if (residual) val += residual[(size_t)gr * N + gc];
      if constexpr (OUT_BF16) {
        ((__bf16*)Cout)[(size_t)gr * N + gc] = (__bf16)val;
      } else {
        ((float*)Cout)[(size_t)gr * N + gc] = val;
      }
    }
  }
}

// ---------------------------------------------------------------------------
// Kernel 2: scores[bh,i,j] = scale * sum_dk q[b,i,h*DK+dk] * k[b,j,h*DK+dk]
// masked to -1e9. q/k are bf16: their 128x64 tiles are DMA'd into LDS by the
// Tensor Data Mover (padded to 72-elem pitch), mask tile rides on ASYNCcnt
// and overlaps the WMMA chain.
// ---------------------------------------------------------------------------
__global__ __launch_bounds__(256)
void scores_kernel(const __bf16* __restrict__ q, const __bf16* __restrict__ k,
                   const unsigned char* __restrict__ mask,
                   float* __restrict__ scores, float scale) {
  constexpr int TP = 72; // 64 data + 8 pad (TDM pad: 4 DWORDs per 32 DWORDs)
  __shared__ __bf16 sQ[128 * TP];
  __shared__ __bf16 sK[128 * TP];
  __shared__ unsigned char smask[128 * 128];

  const int tid  = threadIdx.x;
  const int wave = tid >> 5;
  const int lane = tid & 31;
  const int col  = lane & 15;
  const int half = lane >> 4;
  const int bh = blockIdx.z;
  const int b  = bh >> 4;
  const int h  = bh & 15;
  const int m0 = blockIdx.y * 128;
  const int n0 = blockIdx.x * 128;

  const __bf16* qh = q + (size_t)b * Sc * Dc + h * DKc;
  const __bf16* kh = k + (size_t)b * Sc * Dc + h * DKc;

#if HAVE_TDM
  if (wave == 0) {
    tdm_load_2d_bf16((unsigned)(size_t)&sQ[0], qh + (size_t)m0 * Dc,
                     Dc, (unsigned)(Bc * Sc), Dc, DKc, 128);
    tdm_load_2d_bf16((unsigned)(size_t)&sK[0], kh + (size_t)n0 * Dc,
                     Dc, (unsigned)(Bc * Sc), Dc, DKc, 128);
  }
#else
  // Fallback: per-lane async global->LDS (b128 = 8 bf16 per lane)
#pragma unroll
  for (int t = 0; t < 4; ++t) {
    int idx = t * 256 + tid;
    int r = idx >> 3, c8 = (idx & 7) * 8;
    unsigned lq = (unsigned)(size_t)&sQ[r * TP + c8];
    unsigned lk = (unsigned)(size_t)&sK[r * TP + c8];
    const __bf16* gq = qh + (size_t)(m0 + r) * Dc + c8;
    const __bf16* gk = kh + (size_t)(n0 + r) * Dc + c8;
    asm volatile("global_load_async_to_lds_b128 %0, %1, off" :: "v"(lq), "v"(gq) : "memory");
    asm volatile("global_load_async_to_lds_b128 %0, %1, off" :: "v"(lk), "v"(gk) : "memory");
  }
#endif

  // Mask tile (16 KB of bytes) on ASYNCcnt, issued after q/k so a counted wait
  // can release q/k while the mask DMA still overlaps the WMMAs.
  {
    const unsigned char* mbase = mask + ((size_t)b * Sc + m0) * Sc + n0;
#pragma unroll
    for (int t = 0; t < 4; ++t) {
      int idx = t * 256 + tid;
      int r = idx >> 3, c16 = (idx & 7) * 16;
      unsigned loff = (unsigned)(size_t)&smask[r * 128 + c16];
      const unsigned char* gp = mbase + (size_t)r * Sc + c16;
      asm volatile("global_load_async_to_lds_b128 %0, %1, off"
                   :: "v"(loff), "v"(gp) : "memory");
    }
  }

#if HAVE_TDM
  if (wave == 0) __builtin_amdgcn_s_wait_tensorcnt(0);
#else
  wait_asynccnt4(); // q/k async loads (first 8/wave) complete, mask may fly on
#endif
  __syncthreads();

  v8f acc[8] = {};
#pragma unroll
  for (int ks = 0; ks < DKc; ks += 32) {
    v16bf a = load_a_frag(sQ + (wave * 16 + (lane & 15)) * TP + ks, half);
    v16bf bcur = load_b_frag(sK + col * TP + ks, half);
#pragma unroll
    for (int nf = 0; nf < 8; ++nf) {
      v16bf bnxt = bcur;
      if (nf < 7) bnxt = load_b_frag(sK + ((nf + 1) * 16 + col) * TP + ks, half);
      acc[nf] = __builtin_amdgcn_wmma_f32_16x16x32_bf16(
          false, a, false, bcur, (short)0, acc[nf], false, false);
      bcur = bnxt;
    }
  }

  wait_asynccnt0();   // mask DMA done (own wait + barrier covers all waves)
  __syncthreads();

  const int mrow = m0 + wave * 16;
#pragma unroll
  for (int nf = 0; nf < 8; ++nf) {
    int gj = n0 + nf * 16 + col;
#pragma unroll
    for (int vv = 0; vv < 8; ++vv) {
      int gi = mrow + vv + 8 * half;
      float val = acc[nf][vv] * scale;
      unsigned char mb = smask[(gi - m0) * 128 + (gj - n0)];
      val = mb ? -1e9f : val;
      scores[((size_t)bh * Sc + gi) * Sc + gj] = val;
    }
  }
}

// ---------------------------------------------------------------------------
// Kernel 3: row softmax over 1024 columns. 256-thread block per row, float4
// I/O, wave32 shfl_xor reductions + one 8-entry LDS pass.
// ---------------------------------------------------------------------------
__global__ __launch_bounds__(256)
void softmax_kernel(const float* __restrict__ scores, float* __restrict__ attn) {
  __shared__ float redm[8];
  __shared__ float reds[8];
  const int tid  = threadIdx.x;
  const int wave = tid >> 5;
  const int lane = tid & 31;
  const size_t row = blockIdx.x;

  float4 x = ((const float4*)(scores + row * Sc))[tid];

  float m = fmaxf(fmaxf(x.x, x.y), fmaxf(x.z, x.w));
#pragma unroll
  for (int off = 16; off > 0; off >>= 1) m = fmaxf(m, __shfl_xor(m, off, 32));
  if (lane == 0) redm[wave] = m;
  __syncthreads();
  float rmax = redm[0];
#pragma unroll
  for (int w = 1; w < 8; ++w) rmax = fmaxf(rmax, redm[w]);

  float e0 = __expf(x.x - rmax), e1 = __expf(x.y - rmax);
  float e2 = __expf(x.z - rmax), e3 = __expf(x.w - rmax);
  float s = (e0 + e1) + (e2 + e3);
#pragma unroll
  for (int off = 16; off > 0; off >>= 1) s += __shfl_xor(s, off, 32);
  if (lane == 0) reds[wave] = s;
  __syncthreads();
  float rsum = reds[0];
#pragma unroll
  for (int w = 1; w < 8; ++w) rsum += reds[w];
  float inv = 1.0f / rsum;

  float4 o; o.x = e0 * inv; o.y = e1 * inv; o.z = e2 * inv; o.w = e3 * inv;
  ((float4*)(attn + row * Sc))[tid] = o;
}

// ---------------------------------------------------------------------------
// Kernel 4: ctx[b,i,h*DK+dk] = sum_j attn[bh,i,j] * v[b,j,h*DK+dk]
// Block tile 128(M) x 64(N=DK), K-step 32 over j. bf16 V tile transposed into
// LDS so B-fragment pair loads are contiguous along j.
// ---------------------------------------------------------------------------
__global__ __launch_bounds__(256)
void context_kernel(const float* __restrict__ attn, const __bf16* __restrict__ vproj,
                    float* __restrict__ ctx) {
  constexpr int TP = 40;
  __shared__ __bf16 sA[128 * TP]; // attn tile [row][j]
  __shared__ __bf16 sV[64 * TP];  // v^T tile  [dk][j]

  const int tid  = threadIdx.x;
  const int wave = tid >> 5;
  const int lane = tid & 31;
  const int col  = lane & 15;
  const int half = lane >> 4;
  const int bh = blockIdx.z;
  const int b  = bh >> 4;
  const int h  = bh & 15;
  const int m0 = blockIdx.y * 128;

  const float* ah = attn + (size_t)bh * Sc * Sc;
  const __bf16* vh = vproj + (size_t)b * Sc * Dc + h * DKc;

  v8f acc[4] = {};

  for (int jk = 0; jk < Sc; jk += 32) {
    // attn tile 128x32 fp32 -> packed bf16
#pragma unroll
    for (int i = 0; i < 4; ++i) {
      int idx = i * 256 + tid;
      int r = idx >> 3, c4 = (idx & 7) * 4;
      float4 xa = *(const float4*)&ah[(size_t)(m0 + r) * Sc + jk + c4];
      uint2 pa; pa.x = pack_bf16(xa.x, xa.y); pa.y = pack_bf16(xa.z, xa.w);
      *(uint2*)&sA[r * TP + c4] = pa;
      if (jk + 32 < Sc)
        __builtin_prefetch(&ah[(size_t)(m0 + r) * Sc + jk + 32 + c4], 0, 1);
    }
    // v tile 32(j) x 64(dk) bf16: 16B coalesced reads, transposed LDS writes
    {
      int jr = tid >> 3, dk8 = (tid & 7) * 8;
      union { uint4 u; __bf16 h[8]; } t;
      t.u = *(const uint4*)&vh[(size_t)(jk + jr) * Dc + dk8];
#pragma unroll
      for (int e = 0; e < 8; ++e) sV[(dk8 + e) * TP + jr] = t.h[e];
    }
    __syncthreads();

    v16bf a = load_a_frag(sA + (wave * 16 + (lane & 15)) * TP, half);
    v16bf bcur = load_b_frag(sV + col * TP, half);
#pragma unroll
    for (int nf = 0; nf < 4; ++nf) {
      v16bf bnxt = bcur;
      if (nf < 3) bnxt = load_b_frag(sV + ((nf + 1) * 16 + col) * TP, half);
      acc[nf] = __builtin_amdgcn_wmma_f32_16x16x32_bf16(
          false, a, false, bcur, (short)0, acc[nf], false, false);
      bcur = bnxt;
    }
    __syncthreads();
  }

  const int mrow = m0 + wave * 16;
#pragma unroll
  for (int nf = 0; nf < 4; ++nf) {
    int gdk = nf * 16 + col;
#pragma unroll
    for (int vv = 0; vv < 8; ++vv) {
      int gi = mrow + vv + 8 * half;
      ctx[((size_t)b * Sc + gi) * Dc + h * DKc + gdk] = acc[nf][vv];
    }
  }
}

// ---------------------------------------------------------------------------
// Host-side launch
// ---------------------------------------------------------------------------
extern "C" void kernel_launch(void* const* d_in, const int* in_sizes, int n_in,
                              void* d_out, int out_size, void* d_ws, size_t ws_size,
                              hipStream_t stream) {
  (void)in_sizes; (void)n_in; (void)out_size; (void)ws_size;

  const float* Qin  = (const float*)d_in[0];
  const float* Kin  = (const float*)d_in[1];
  const float* Vin  = (const float*)d_in[2];
  const unsigned char* maskp = (const unsigned char*)d_in[3];
  const float* Wq = (const float*)d_in[4];
  const float* bq = (const float*)d_in[5];
  const float* Wk = (const float*)d_in[6];
  const float* bk = (const float*)d_in[7];
  const float* Wv = (const float*)d_in[8];
  const float* bv = (const float*)d_in[9];
  const float* Wo = (const float*)d_in[10];
  const float* bo = (const float*)d_in[11];

  const size_t BSD  = (size_t)Bc * Sc * Dc;       // 8M elements
  const size_t BHSS = (size_t)Bc * Hc * Sc * Sc;  // 128M elements

  __bf16* qb = (__bf16*)d_ws;          // projections kept in bf16
  __bf16* kb = qb + BSD;
  __bf16* vb = kb + BSD;
  float*  ctxp = (float*)(vb + BSD);   // context in fp32

  float* outp    = (float*)d_out;  // [B,S,D]
  float* attnp   = outp + BSD;     // [B,H,S,S]
  float* scoresp = attnp + BHSS;   // [B,H,S,S]

  const int M = Bc * Sc, N = Dc, K = Dc;

  dim3 blk(256);
  dim3 gProj(N / 128, M / 128);  // 8 x 64

  proj_gemm_kernel<true><<<gProj, blk, 0, stream>>>(Qin, Wq, bq, nullptr, qb, M, N, K);
  proj_gemm_kernel<true><<<gProj, blk, 0, stream>>>(Kin, Wk, bk, nullptr, kb, M, N, K);
  proj_gemm_kernel<true><<<gProj, blk, 0, stream>>>(Vin, Wv, bv, nullptr, vb, M, N, K);

  dim3 gScores(Sc / 128, Sc / 128, Bc * Hc);  // 8 x 8 x 128
  scores_kernel<<<gScores, blk, 0, stream>>>(qb, kb, maskp, scoresp, 0.125f);

  dim3 gSoft((unsigned)(Bc * Hc * Sc));  // 131072 rows
  softmax_kernel<<<gSoft, blk, 0, stream>>>(scoresp, attnp);

  dim3 gCtx(1, Sc / 128, Bc * Hc);  // 1 x 8 x 128
  context_kernel<<<gCtx, blk, 0, stream>>>(attnp, vb, ctxp);

  proj_gemm_kernel<false><<<gProj, blk, 0, stream>>>(ctxp, Wo, bo, Qin, outp, M, N, K);
}
